// GRUDCell_34333968564821
// MI455X (gfx1250) — matrix operand order repaired and checked by reading, not compiled, not run on hardware
//
#include <hip/hip_runtime.h>
#include <hip/hip_bf16.h>

#define BB   16384
#define IN_  512
#define HH   1024
#define DD   512
#define BKP  40   // LDS pitch in bf16 elements: 64B row + 16B pad = 80B (bank-conflict free)

typedef __attribute__((ext_vector_type(16))) __bf16 v16bf;
typedef __attribute__((ext_vector_type(8)))  float  v8f;
typedef __attribute__((ext_vector_type(4)))  unsigned int u32x4;
typedef __attribute__((ext_vector_type(8)))  unsigned int u32x8;

union Frag { uint4 u4[2]; v16bf v; };

__device__ __forceinline__ v8f wmma_bf16(const Frag& a, const Frag& b, v8f c) {
  return __builtin_amdgcn_wmma_f32_16x16x32_bf16(
      /*neg_a=*/false, a.v, /*neg_b=*/false, b.v,
      /*c_mod=*/(short)0, c, /*reuse_a=*/false, /*reuse_b=*/false);
}

__device__ __forceinline__ unsigned short f2bf(float x) {
  unsigned int u = __float_as_uint(x);
  return (unsigned short)((u + 0x7FFFu + ((u >> 16) & 1u)) >> 16);  // RNE
}

__device__ __forceinline__ float sigmoid_f(float x) {
  return 1.0f / (1.0f + __expf(-x));
}
__device__ __forceinline__ float tanh_f(float x) {
  return 2.0f * sigmoid_f(2.0f * x) - 1.0f;
}

// -------- CDNA5 Tensor Data Mover: one instruction moves a whole 2D tile to LDS --------
// D# groups built in SGPRs (all inputs uniform). Tile: rows x 32 bf16, row-major with
// tensor_dim0_stride = ldk. LDS destination padded per row: 64B data + 16B pad (BKP=40).
__device__ __forceinline__ void tdm_load_tile(
    const unsigned short* g, int ldk, int row0, int k0,
    unsigned short* lds, int rows) {
  unsigned long long ga =
      (unsigned long long)(uintptr_t)(g + (size_t)row0 * ldk + k0);
  u32x4 g0;
  g0[0] = 1u;                                         // count=1, user mode
  g0[1] = (unsigned)(uintptr_t)lds;                   // lds_addr (byte offset)
  g0[2] = (unsigned)ga;                               // global_addr[31:0]
  g0[3] = (unsigned)((ga >> 32) & 0x01FFFFFFu) | (2u << 30);  // addr[56:32] | type=2
  u32x8 g1;
  g1[0] = (1u << 16)      // data_size = 2 bytes
        | (1u << 20)      // pad_enable
        | (3u << 22)      // pad_interval: every 64 bytes
        | (3u << 25);     // pad_amount: 4 DWORDs = 16 bytes
  unsigned d0 = (unsigned)ldk;    // tensor_dim0 (elements)
  unsigned d1 = (unsigned)rows;   // tensor_dim1
  g1[1] = (d0 & 0xFFFFu) << 16;                       // tensor_dim0[15:0]
  g1[2] = (d0 >> 16) | ((d1 & 0xFFFFu) << 16);        // tensor_dim0[31:16] | tensor_dim1[15:0]
  g1[3] = (d1 >> 16) | (32u << 16);                   // tensor_dim1[31:16] | tile_dim0=32
  g1[4] = (unsigned)rows;                             // tile_dim1=rows, tile_dim2=0
  g1[5] = (unsigned)ldk;                              // tensor_dim0_stride[31:0]
  g1[6] = 0u;
  g1[7] = 0u;
  asm volatile("tensor_load_to_lds %0, %1" :: "s"(g0), "s"(g1) : "memory");
}
__device__ __forceinline__ void wait_tensor0() {
  __builtin_amdgcn_s_wait_tensorcnt(0);
}

// ---------------- f32 -> bf16 conversion ----------------
__global__ __launch_bounds__(256) void convert_bf16_kernel(
    const float* __restrict__ src, unsigned short* __restrict__ dst, int n) {
  for (int i = blockIdx.x * blockDim.x + threadIdx.x; i < n;
       i += gridDim.x * blockDim.x)
    dst[i] = f2bf(src[i]);
}

// A fragment: 16x32 bf16. lane l: M = l&15; half h = l>>4 holds K = h*8..h*8+7 and 16+h*8..+7
__device__ __forceinline__ Frag load_a_frag(const unsigned short* lds, int rowBase, int lane) {
  Frag f;
  int m = lane & 15, kh = (lane >> 4) << 3;
  const unsigned short* p = lds + (rowBase + m) * BKP;
  f.u4[0] = *(const uint4*)(p + kh);
  f.u4[1] = *(const uint4*)(p + 16 + kh);
  return f;
}

// B fragment: 32x16 bf16 (B[k][n] = W[n][k], W row-major). lane l: N = l&15; half h = l>>4 holds K = h*8..h*8+7 and 16+h*8..+7
__device__ __forceinline__ Frag load_b_frag(const unsigned short* lds, int nBase, int lane) {
  Frag f;
  int n = lane & 15, kh = (lane >> 4) << 3;
  const unsigned short* p = lds + (nBase + n) * BKP + kh;
  f.u4[0] = *(const uint4*)(p);
  f.u4[1] = *(const uint4*)(p + 16);
  return f;
}

// ---------------- Stage 1: h_dec = exp(-relu(delta @ Wdh.T + bdh)) * h ----------------
__global__ __launch_bounds__(256) void decay_h_kernel(
    const unsigned short* __restrict__ dbf,   // [B, D] bf16
    const unsigned short* __restrict__ wbf,   // [H, D] bf16
    const float* __restrict__ bdh, const float* __restrict__ h,
    float* __restrict__ hdec_f, unsigned short* __restrict__ hdec_bf) {
  __shared__ __align__(16) unsigned short sA[2][128 * BKP];
  __shared__ __align__(16) unsigned short sB[2][64 * BKP];
  int tid = threadIdx.x, lane = tid & 31, wave = tid >> 5;
  int wm = wave >> 1, wn = wave & 1;
  int row0 = blockIdx.x * 128, col0 = blockIdx.y * 64;
  v8f z = {};
  v8f acc[2][2] = {{z, z}, {z, z}};
  const int nk = DD / 32;
  if (wave == 0) {
    tdm_load_tile(dbf, DD, row0, 0, sA[0], 128);
    tdm_load_tile(wbf, DD, col0, 0, sB[0], 64);
  }
  for (int i = 0; i < nk; ++i) {
    int cur = i & 1, nxt = cur ^ 1;
    if (wave == 0) wait_tensor0();
    __syncthreads();
    if (wave == 0 && i + 1 < nk) {
      tdm_load_tile(dbf, DD, row0, (i + 1) * 32, sA[nxt], 128);
      tdm_load_tile(wbf, DD, col0, (i + 1) * 32, sB[nxt], 64);
    }
    Frag a0 = load_a_frag(sA[cur], wm * 32, lane);
    Frag a1 = load_a_frag(sA[cur], wm * 32 + 16, lane);
    Frag b0 = load_b_frag(sB[cur], wn * 32, lane);
    Frag b1 = load_b_frag(sB[cur], wn * 32 + 16, lane);
    acc[0][0] = wmma_bf16(a0, b0, acc[0][0]);
    acc[0][1] = wmma_bf16(a0, b1, acc[0][1]);
    acc[1][0] = wmma_bf16(a1, b0, acc[1][0]);
    acc[1][1] = wmma_bf16(a1, b1, acc[1][1]);
  }
  int nl = lane & 15, kh = lane >> 4;
#pragma unroll
  for (int fm = 0; fm < 2; ++fm)
#pragma unroll
    for (int fn = 0; fn < 2; ++fn) {
      int nCol = col0 + wn * 32 + fn * 16 + nl;
      int mBase = row0 + wm * 32 + fm * 16 + kh * 8;
      float bias = bdh[nCol];
#pragma unroll
      for (int j = 0; j < 8; ++j) {
        int m = mBase + j;
        float v = acc[fm][fn][j] + bias;
        float g = __expf(-fmaxf(v, 0.0f));
        size_t idx = (size_t)m * HH + nCol;
        float hd = g * h[idx];
        hdec_f[idx] = hd;
        hdec_bf[idx] = f2bf(hd);
      }
    }
}

// ---------------- Stage 2: x_input = mask*x + (1-mask)*exp(-relu(x @ Wdx.T + bdx))*x ----------------
__global__ __launch_bounds__(256) void decay_x_kernel(
    const unsigned short* __restrict__ xbf,   // [B, IN] bf16
    const unsigned short* __restrict__ wbf,   // [IN, IN] bf16
    const float* __restrict__ bdx, const float* __restrict__ x,
    const float* __restrict__ mask, unsigned short* __restrict__ xi_bf) {
  __shared__ __align__(16) unsigned short sA[2][128 * BKP];
  __shared__ __align__(16) unsigned short sB[2][64 * BKP];
  int tid = threadIdx.x, lane = tid & 31, wave = tid >> 5;
  int wm = wave >> 1, wn = wave & 1;
  int row0 = blockIdx.x * 128, col0 = blockIdx.y * 64;
  v8f z = {};
  v8f acc[2][2] = {{z, z}, {z, z}};
  const int nk = IN_ / 32;
  if (wave == 0) {
    tdm_load_tile(xbf, IN_, row0, 0, sA[0], 128);
    tdm_load_tile(wbf, IN_, col0, 0, sB[0], 64);
  }
  for (int i = 0; i < nk; ++i) {
    int cur = i & 1, nxt = cur ^ 1;
    if (wave == 0) wait_tensor0();
    __syncthreads();
    if (wave == 0 && i + 1 < nk) {
      tdm_load_tile(xbf, IN_, row0, (i + 1) * 32, sA[nxt], 128);
      tdm_load_tile(wbf, IN_, col0, (i + 1) * 32, sB[nxt], 64);
    }
    Frag a0 = load_a_frag(sA[cur], wm * 32, lane);
    Frag a1 = load_a_frag(sA[cur], wm * 32 + 16, lane);
    Frag b0 = load_b_frag(sB[cur], wn * 32, lane);
    Frag b1 = load_b_frag(sB[cur], wn * 32 + 16, lane);
    acc[0][0] = wmma_bf16(a0, b0, acc[0][0]);
    acc[0][1] = wmma_bf16(a0, b1, acc[0][1]);
    acc[1][0] = wmma_bf16(a1, b0, acc[1][0]);
    acc[1][1] = wmma_bf16(a1, b1, acc[1][1]);
  }
  int nl = lane & 15, kh = lane >> 4;
#pragma unroll
  for (int fm = 0; fm < 2; ++fm)
#pragma unroll
    for (int fn = 0; fn < 2; ++fn) {
      int nCol = col0 + wn * 32 + fn * 16 + nl;
      int mBase = row0 + wm * 32 + fm * 16 + kh * 8;
      float bias = bdx[nCol];
#pragma unroll
      for (int j = 0; j < 8; ++j) {
        int m = mBase + j;
        float v = acc[fm][fn][j] + bias;
        float g = __expf(-fmaxf(v, 0.0f));
        size_t idx = (size_t)m * IN_ + nCol;
        float xv = x[idx], mk = mask[idx];
        xi_bf[idx] = f2bf(mk * xv + (1.0f - mk) * g * xv);
      }
    }
}

// ---------------- Stage 3: fused gates + output blend ----------------
__device__ __forceinline__ void gates_issue_tdm(
    int t, const unsigned short* __restrict__ xi_bf,
    const unsigned short* __restrict__ hdec_bf,
    const unsigned short* __restrict__ Wih_bf,
    const unsigned short* __restrict__ Whh_bf,
    int row0, int col0,
    unsigned short* sAb, unsigned short* sB0, unsigned short* sB1, unsigned short* sB2) {
  if (t < IN_ / 32) {
    int k0 = t * 32;
    tdm_load_tile(xi_bf, IN_, row0, k0, sAb, 128);
    tdm_load_tile(Wih_bf, IN_, 0 * HH + col0, k0, sB0, 32);
    tdm_load_tile(Wih_bf, IN_, 1 * HH + col0, k0, sB1, 32);
    tdm_load_tile(Wih_bf, IN_, 2 * HH + col0, k0, sB2, 32);
  } else {
    int k0 = (t - IN_ / 32) * 32;
    tdm_load_tile(hdec_bf, HH, row0, k0, sAb, 128);
    tdm_load_tile(Whh_bf, HH, 0 * HH + col0, k0, sB0, 32);
    tdm_load_tile(Whh_bf, HH, 1 * HH + col0, k0, sB1, 32);
    tdm_load_tile(Whh_bf, HH, 2 * HH + col0, k0, sB2, 32);
  }
}

__global__ __launch_bounds__(256) void gates_kernel(
    const unsigned short* __restrict__ xi_bf,    // [B, IN]
    const unsigned short* __restrict__ hdec_bf,  // [B, H]
    const unsigned short* __restrict__ Wih_bf,   // [3H, IN]
    const unsigned short* __restrict__ Whh_bf,   // [3H, H]
    const float* __restrict__ bih, const float* __restrict__ bhh,
    const float* __restrict__ hdec_f, float* __restrict__ out) {
  __shared__ __align__(16) unsigned short sA[2][128 * BKP];
  __shared__ __align__(16) unsigned short sB[2][3][32 * BKP];
  int tid = threadIdx.x, lane = tid & 31, wave = tid >> 5;
  int wm = wave >> 1, wn = wave & 1;  // 4 waves along M, 2 along N (tile 128x32)
  int row0 = blockIdx.x * 128, col0 = blockIdx.y * 32;
  v8f z = {};
  v8f accR[2] = {z, z}, accU[2] = {z, z}, accN[2] = {z, z}, accHN[2] = {z, z};

  const int NT = IN_ / 32 + HH / 32;  // 16 + 32 = 48 K-tiles, single pipelined loop
  if (wave == 0)
    gates_issue_tdm(0, xi_bf, hdec_bf, Wih_bf, Whh_bf, row0, col0,
                    sA[0], sB[0][0], sB[0][1], sB[0][2]);
  for (int t = 0; t < NT; ++t) {
    int cur = t & 1, nxt = cur ^ 1;
    if (wave == 0) wait_tensor0();
    __syncthreads();
    if (wave == 0 && t + 1 < NT)
      gates_issue_tdm(t + 1, xi_bf, hdec_bf, Wih_bf, Whh_bf, row0, col0,
                      sA[nxt], sB[nxt][0], sB[nxt][1], sB[nxt][2]);
    Frag a0 = load_a_frag(sA[cur], wm * 32, lane);
    Frag a1 = load_a_frag(sA[cur], wm * 32 + 16, lane);
    Frag br = load_b_frag(sB[cur][0], wn * 16, lane);
    Frag bu = load_b_frag(sB[cur][1], wn * 16, lane);
    Frag bn = load_b_frag(sB[cur][2], wn * 16, lane);
    accR[0] = wmma_bf16(a0, br, accR[0]);  accR[1] = wmma_bf16(a1, br, accR[1]);
    accU[0] = wmma_bf16(a0, bu, accU[0]);  accU[1] = wmma_bf16(a1, bu, accU[1]);
    if (t < IN_ / 32) {
      accN[0] = wmma_bf16(a0, bn, accN[0]);  accN[1] = wmma_bf16(a1, bn, accN[1]);
    } else {
      accHN[0] = wmma_bf16(a0, bn, accHN[0]);  accHN[1] = wmma_bf16(a1, bn, accHN[1]);
    }
  }

  int nl = lane & 15, kh = lane >> 4;
  int nCol = col0 + wn * 16 + nl;
  float br_  = bih[nCol] + bhh[nCol];
  float bu_  = bih[HH + nCol] + bhh[HH + nCol];
  float bin_ = bih[2 * HH + nCol];
  float bhn_ = bhh[2 * HH + nCol];
#pragma unroll
  for (int fm = 0; fm < 2; ++fm) {
    int mBase = row0 + wm * 32 + fm * 16 + kh * 8;
#pragma unroll
    for (int j = 0; j < 8; ++j) {
      int m = mBase + j;
      float r  = sigmoid_f(accR[fm][j] + br_);
      float u  = sigmoid_f(accU[fm][j] + bu_);
      float nn = tanh_f(accN[fm][j] + bin_ + r * (accHN[fm][j] + bhn_));
      size_t idx = (size_t)m * HH + nCol;
      float hd = hdec_f[idx];
      float hy = (1.0f - u) * nn + u * hd;
      out[idx] = hy;
      out[(size_t)BB * HH + idx] = hy;  // tuple (hy, hy)
    }
  }
}

extern "C" void kernel_launch(void* const* d_in, const int* in_sizes, int n_in,
                              void* d_out, int out_size, void* d_ws, size_t ws_size,
                              hipStream_t stream) {
  (void)in_sizes; (void)n_in; (void)out_size; (void)ws_size;
  const float* x     = (const float*)d_in[0];
  const float* h     = (const float*)d_in[1];
  const float* mask  = (const float*)d_in[2];
  const float* delta = (const float*)d_in[3];
  const float* Wih   = (const float*)d_in[4];
  const float* bih   = (const float*)d_in[5];
  const float* Whh   = (const float*)d_in[6];
  const float* bhh   = (const float*)d_in[7];
  const float* Wdh   = (const float*)d_in[8];
  const float* bdh   = (const float*)d_in[9];
  const float* Wdx   = (const float*)d_in[10];
  const float* bdx   = (const float*)d_in[11];
  float* out = (float*)d_out;
  char* ws = (char*)d_ws;

  size_t o = 0;
  auto alloc = [&](size_t bytes) {
    size_t r = o; o += (bytes + 255) & ~(size_t)255; return r;
  };
  unsigned short* delta_bf = (unsigned short*)(ws + alloc((size_t)BB * DD * 2));
  unsigned short* x_bf     = (unsigned short*)(ws + alloc((size_t)BB * IN_ * 2));
  unsigned short* Wdh_bf   = (unsigned short*)(ws + alloc((size_t)HH * DD * 2));
  unsigned short* Wdx_bf   = (unsigned short*)(ws + alloc((size_t)IN_ * IN_ * 2));
  unsigned short* Wih_bf   = (unsigned short*)(ws + alloc((size_t)3 * HH * IN_ * 2));
  unsigned short* Whh_bf   = (unsigned short*)(ws + alloc((size_t)3 * HH * HH * 2));
  float*          hdec_f   = (float*)(ws + alloc((size_t)BB * HH * 4));
  unsigned short* hdec_bf  = (unsigned short*)(ws + alloc((size_t)BB * HH * 2));
  unsigned short* xi_bf    = (unsigned short*)(ws + alloc((size_t)BB * IN_ * 2));

  auto cvt = [&](const float* s, unsigned short* d, int n) {
    int blocks = (n + 255) / 256; if (blocks > 4096) blocks = 4096;
    convert_bf16_kernel<<<dim3(blocks), dim3(256), 0, stream>>>(s, d, n);
  };
  cvt(delta, delta_bf, BB * DD);
  cvt(x,     x_bf,     BB * IN_);
  cvt(Wdh,   Wdh_bf,   HH * DD);
  cvt(Wdx,   Wdx_bf,   IN_ * IN_);
  cvt(Wih,   Wih_bf,   3 * HH * IN_);
  cvt(Whh,   Whh_bf,   3 * HH * HH);

  decay_h_kernel<<<dim3(BB / 128, HH / 64), 256, 0, stream>>>(
      delta_bf, Wdh_bf, bdh, h, hdec_f, hdec_bf);
  decay_x_kernel<<<dim3(BB / 128, IN_ / 64), 256, 0, stream>>>(
      x_bf, Wdx_bf, bdx, x, mask, xi_bf);
  gates_kernel<<<dim3(BB / 128, HH / 32), 256, 0, stream>>>(
      xi_bf, hdec_bf, Wih_bf, Whh_bf, bih, bhh, hdec_f, out);
}
